// HeteroDictLinear_40286793237062
// MI455X (gfx1250) — matrix operand order repaired and checked by reading, not compiled, not run on hardware
//
#include <hip/hip_runtime.h>
#include <hip/hip_bf16.h>

// ---------------------------------------------------------------------------
// HeteroDictLinear on MI455X (gfx1250, wave32, WMMA)
//
// y_k = x_k @ W_k^T + b_k for 6 node types. All OUT=128.
// Memory-bound problem (~768 MB @ 23.3 TB/s => ~33us floor). fp32 WMMA
// (16x16x4) would be compute-bound, so we split fp32 into bf16 hi+lo and use
// V_WMMA_F32_16X16X32_BF16 three times per tile (ah*bh + al*bh + ah*bl) with
// fp32 accumulation: near-fp32 accuracy at bf16 matrix-engine rate.
// ---------------------------------------------------------------------------

typedef __attribute__((ext_vector_type(16))) __bf16 v16bf;
typedef __attribute__((ext_vector_type(8)))  __bf16 v8bf;
typedef __attribute__((ext_vector_type(8)))  float  v8f;
typedef __attribute__((ext_vector_type(4)))  float  v4f;   // native clang vector
                                                           // (HIP float4 is a
                                                           // struct; NT builtins
                                                           // reject it)

#define OUT_N 128

__device__ __forceinline__ __bf16 bf_hi(float f) { return (__bf16)f; }
__device__ __forceinline__ __bf16 bf_lo(float f, __bf16 hi) {
    return (__bf16)(f - (float)hi);
}

template <int K>
__global__ __launch_bounds__(256)
void hetero_linear_kernel(const float* __restrict__ X,
                          const float* __restrict__ W,
                          const float* __restrict__ Bias,
                          float* __restrict__ Y,
                          int nrows)
{
    // Padded LDS row stride: +8 bf16 elems = +16B => row stride mod 64 banks
    // is 4 dwords, so the 16 lanes of a column read hit 16 distinct banks.
    constexpr int KP = K + 8;
    extern __shared__ __bf16 lds[];
    __bf16* w_hi = lds;                 // [OUT_N][KP] bf16 hi parts
    __bf16* w_lo = lds + OUT_N * KP;    // [OUT_N][KP] bf16 lo parts

    const int tid = threadIdx.x;

    // ---- Stage W into LDS as bf16 hi/lo (cooperative, once per block) ----
    for (int i = tid; i < OUT_N * K; i += 256) {
        const int r = i / K;
        const int c = i - r * K;
        const float w  = W[i];
        const __bf16 h = bf_hi(w);
        w_hi[r * KP + c] = h;
        w_lo[r * KP + c] = bf_lo(w, h);
    }
    __syncthreads();

    const int wave = tid >> 5;          // 0..7, each wave owns 16 rows
    const int lane = tid & 31;
    const int lrow = lane & 15;         // row-in-tile (A) / column (B,C,D)
    const int ksel = lane >> 4;         // half-wave K interleave selector

    const long row0 = (long)blockIdx.x * 128 + (long)wave * 16;

    // Clamp A-row for loads (stores are predicated; WMMA needs EXEC all-1s,
    // so no early returns anywhere).
    long arow = row0 + lrow;
    if (arow > (long)(nrows - 1)) arow = (long)(nrows - 1);
    const float* __restrict__ xrow = X + (size_t)arow * K;

    // ---- Accumulators: 8 N-tiles of 16x16, init with bias ----
    v8f acc[8];
#pragma unroll
    for (int t = 0; t < 8; ++t) {
        const float bv = Bias[t * 16 + lrow];
#pragma unroll
        for (int r = 0; r < 8; ++r) acc[t][r] = bv;
    }

    // ---- K loop: chunks of 32 (one bf16 WMMA depth) ----
#pragma unroll
    for (int kc = 0; kc < K; kc += 32) {
        // A fragment (16x32 bf16, contiguous layout matching B):
        //  lanes 0-15 : row=lrow, elems 0..15 = K kc+0..15
        //  lanes16-31 : row=lrow, elems 0..15 = K kc+16..31
        const float* p = xrow + kc + ksel * 16;
        const v4f s0 = __builtin_nontemporal_load((const v4f*)(p + 0));
        const v4f s1 = __builtin_nontemporal_load((const v4f*)(p + 4));
        const v4f s2 = __builtin_nontemporal_load((const v4f*)(p + 8));
        const v4f s3 = __builtin_nontemporal_load((const v4f*)(p + 12));
        const float a32[16] = { s0[0], s0[1], s0[2], s0[3],
                                s1[0], s1[1], s1[2], s1[3],
                                s2[0], s2[1], s2[2], s2[3],
                                s3[0], s3[1], s3[2], s3[3] };
        v16bf a_hi, a_lo;
#pragma unroll
        for (int j = 0; j < 16; ++j) {
            const __bf16 h = bf_hi(a32[j]);
            a_hi[j] = h;
            a_lo[j] = bf_lo(a32[j], h);
        }

        // B fragments from LDS (32x16 bf16): lane holds column n = t*16+lrow,
        // lanes 0-15 carry K kc+0..15, lanes 16-31 carry K kc+16..31
        // (contiguous 32B per lane => two ds_load_b128).
#pragma unroll
        for (int t = 0; t < 8; ++t) {
            const int n   = t * 16 + lrow;
            const int off = n * KP + kc + ksel * 16;

            const v8bf h0 = *(const v8bf*)(w_hi + off);
            const v8bf h1 = *(const v8bf*)(w_hi + off + 8);
            const v8bf l0 = *(const v8bf*)(w_lo + off);
            const v8bf l1 = *(const v8bf*)(w_lo + off + 8);
            const v16bf b_hi = __builtin_shufflevector(
                h0, h1, 0,1,2,3,4,5,6,7,8,9,10,11,12,13,14,15);
            const v16bf b_lo = __builtin_shufflevector(
                l0, l1, 0,1,2,3,4,5,6,7,8,9,10,11,12,13,14,15);

            // 3-term bf16 split ~ fp32:  c += ah*bh + al*bh + ah*bl
            acc[t] = __builtin_amdgcn_wmma_f32_16x16x32_bf16(
                false, a_hi, false, b_hi, (short)0, acc[t], false, false);
            acc[t] = __builtin_amdgcn_wmma_f32_16x16x32_bf16(
                false, a_lo, false, b_hi, (short)0, acc[t], false, false);
            acc[t] = __builtin_amdgcn_wmma_f32_16x16x32_bf16(
                false, a_hi, false, b_lo, (short)0, acc[t], false, false);
        }
    }

    // ---- Writeback: C/D layout => VGPR r holds row (ksel*8 + r), col lrow ----
#pragma unroll
    for (int r = 0; r < 8; ++r) {
        const long m = row0 + ksel * 8 + r;
        if (m < (long)nrows) {
            float* yrow = Y + (size_t)m * OUT_N;
#pragma unroll
            for (int t = 0; t < 8; ++t)
                __builtin_nontemporal_store(acc[t][r], yrow + t * 16 + lrow);
        }
    }
}

template <int K>
static void launch_one(const float* X, const float* W, const float* B,
                       float* Y, int n, hipStream_t stream)
{
    constexpr int KP = K + 8;
    const size_t shmem = (size_t)2 * OUT_N * KP * sizeof(__bf16); // hi+lo
    // CDNA5 allows up to 320KB LDS per workgroup; raise the runtime cap.
    (void)hipFuncSetAttribute(
        reinterpret_cast<const void*>(&hetero_linear_kernel<K>),
        hipFuncAttributeMaxDynamicSharedMemorySize, (int)shmem);
    const int blocks = (n + 127) / 128;
    hetero_linear_kernel<K><<<blocks, 256, shmem, stream>>>(X, W, B, Y, n);
}

extern "C" void kernel_launch(void* const* d_in, const int* in_sizes, int n_in,
                              void* d_out, int out_size, void* d_ws, size_t ws_size,
                              hipStream_t stream)
{
    (void)in_sizes; (void)n_in; (void)out_size; (void)d_ws; (void)ws_size;
    // setup_inputs order: per type {x, W, b}:
    //   author(100000,128) paper(250000,256) venue(25000,64)
    //   institution(50000,128) field(75000,64) term(150000,128)
    static const int NS[6] = {100000, 250000, 25000, 50000, 75000, 150000};
    static const int KS[6] = {128, 256, 64, 128, 64, 128};

    long off = 0;
    for (int i = 0; i < 6; ++i) {
        const float* X = (const float*)d_in[3 * i + 0];
        const float* W = (const float*)d_in[3 * i + 1];
        const float* B = (const float*)d_in[3 * i + 2];
        float* Y = (float*)d_out + off;
        switch (KS[i]) {
            case 64:  launch_one<64 >(X, W, B, Y, NS[i], stream); break;
            case 128: launch_one<128>(X, W, B, Y, NS[i], stream); break;
            case 256: launch_one<256>(X, W, B, Y, NS[i], stream); break;
        }
        off += (long)NS[i] * OUT_N;
    }
}
